// VLAD_57415122813587
// MI455X (gfx1250) — matrix operand order repaired and checked by reading, not compiled, not run on hardware
//
#include <hip/hip_runtime.h>
#include <hip/hip_bf16.h>

#define B_ 16
#define N_ 8192
#define D_ 128
#define K_ 128

#define ROWS_PER_BLOCK 1024
#define TILES_PER_BLOCK (ROWS_PER_BLOCK / 16)   // 64
#define WAVES 8

typedef __attribute__((ext_vector_type(16))) __bf16 v16bf;
typedef __attribute__((ext_vector_type(8)))  float  v8f;

// ---------------- kernel 0: zero accumulators ----------------
__global__ void vlad_zero(float* __restrict__ p, int n) {
    int i = blockIdx.x * blockDim.x + threadIdx.x;
    if (i < n) p[i] = 0.0f;
}

// ---------------- kernel 1: centroids + c2 ----------------
__global__ __launch_bounds__(256)
void vlad_prep(const float* __restrict__ cs, const float* __restrict__ pop,
               float* __restrict__ centroids, float* __restrict__ c2) {
    int tid = threadIdx.x;
    for (int i = tid; i < K_ * D_; i += 256) {
        int k = i >> 7;
        centroids[i] = cs[i] / pop[k];
    }
    __syncthreads();
    if (tid < K_) {
        float s = 0.0f;
        for (int d = 0; d < D_; ++d) {
            float v = centroids[tid * D_ + d];
            s += v * v;
        }
        c2[tid] = s;
    }
}

// ---------------- kernel 2: assign + accumulate ----------------
// LDS layout (dynamic):
//   accum      @ 0        : 128*128 f32 = 65536 B
//   popsL      @ 65536    : 128 f32     = 512 B
//   Bstage     @ 66048    : 8c*4kt*32ln*16 bf16 = 32768 B (per-lane WMMA B layout)
//   c2L        @ 98816    : 128 f32     = 512 B
//   scores     @ 99328    : 8 waves * 16*128 f32 = 65536 B
//   clusterIds @ 164864   : 8*16 int    = 512 B
__global__ __launch_bounds__(256)
void vlad_assign(const float* __restrict__ descs, const float* __restrict__ centroids,
                 const float* __restrict__ c2, float* __restrict__ desc_sums,
                 float* __restrict__ pops) {
    extern __shared__ char smem[];
    float*  accum      = (float*)(smem);
    float*  popsL      = (float*)(smem + 65536);
    __bf16* Bst        = (__bf16*)(smem + 66048);
    float*  c2L        = (float*)(smem + 98816);
    float*  scores     = (float*)(smem + 99328);
    int*    clusterIds = (int*)(smem + 164864);

    const int tid  = threadIdx.x;
    const int lane = tid & 31;
    const int wave = tid >> 5;
    const int blocksPerBatch = N_ / ROWS_PER_BLOCK;
    const int b       = blockIdx.x / blocksPerBatch;
    const int chunk   = blockIdx.x % blocksPerBatch;
    const int rowBase = chunk * ROWS_PER_BLOCK;

    for (int i = tid; i < K_ * D_; i += 256) accum[i] = 0.0f;
    if (tid < K_) { popsL[tid] = 0.0f; c2L[tid] = c2[tid]; }

    // Stage B (centroids, bf16) in per-lane layout, contiguous k order matching A:
    // B is 32x16 per (c,kt): VGPR j-pair; lanes 0-15 hold K=0..15, lanes 16-31 hold K=16..31
    for (int e = tid; e < 8 * 4 * 32; e += 256) {
        int ln = e & 31;
        int kt = (e >> 5) & 3;
        int c  = e >> 7;
        int ncol  = ln & 15;
        int kbase = (ln >> 4) * 16;
        __bf16* dst = Bst + (size_t)e * 16;
        #pragma unroll
        for (int j = 0; j < 16; ++j) {
            int dim = kt * 32 + kbase + j;
            dst[j] = (__bf16)centroids[(c * 16 + ncol) * D_ + dim];
        }
    }
    __syncthreads();

    const int row_in = lane & 15;
    const int kb     = (lane >> 4) * 16;  // A-matrix K base per half-wave

    for (int t = wave; t < TILES_PER_BLOCK; t += WAVES) {
        const int row0 = rowBase + t * 16;
        const float* drow = descs + ((size_t)b * N_ + row0 + row_in) * D_;

        v8f acc[8] = {};

        #pragma unroll
        for (int kt = 0; kt < 4; ++kt) {
            // Build A operand (16x32 bf16, per-lane layout)
            v16bf a;
            #pragma unroll
            for (int j = 0; j < 16; ++j) {
                int dim = kt * 32 + kb + j;
                a[j] = (__bf16)drow[dim];
            }
            #pragma unroll
            for (int c = 0; c < 8; ++c) {
                v16bf bmat = *((const v16bf*)(Bst + ((size_t)((c * 4 + kt) * 32 + lane)) * 16));
                acc[c] = __builtin_amdgcn_wmma_f32_16x16x32_bf16(
                    false, a, false, bmat, (short)0, acc[c], false, false);
            }
        }

        // Scores: dist-rank = c2[col] - 2*xc ; C tile layout: VGPR i -> M = i + 8*(lane>=16)
        float* sw = scores + wave * (16 * K_);
        const int mtop = (lane >> 4) * 8;
        #pragma unroll
        for (int c = 0; c < 8; ++c) {
            int col  = c * 16 + (lane & 15);
            float cc = c2L[col];
            #pragma unroll
            for (int i = 0; i < 8; ++i) {
                sw[(mtop + i) * K_ + col] = cc - 2.0f * acc[c][i];
            }
        }
        __syncthreads();

        // Argmin per row (lanes 0-15 handle one row each)
        if (lane < 16) {
            const float* srow = sw + lane * K_;
            int best = 0; float bv = srow[0];
            for (int c = 1; c < K_; ++c) {
                float v = srow[c];
                if (v < bv) { bv = v; best = c; }
            }
            clusterIds[wave * 16 + lane] = best;
        }
        __syncthreads();

        // LDS scatter-accumulate descs (fp32) into accum[cluster]
        #pragma unroll 4
        for (int m = 0; m < 16; ++m) {
            int k = clusterIds[wave * 16 + m];
            const float* r = descs + ((size_t)b * N_ + row0 + m) * D_;
            #pragma unroll
            for (int q = 0; q < 4; ++q) {
                int d = lane + 32 * q;
                atomicAdd(&accum[k * D_ + d], r[d]);
            }
        }
        if (lane < 16) atomicAdd(&popsL[clusterIds[wave * 16 + lane]], 1.0f);
        __syncthreads();
    }

    __syncthreads();
    // Merge block-local accumulators into global (L2-resident, 1 MB region)
    for (int i = tid; i < K_ * D_; i += 256)
        atomicAdd(&desc_sums[(size_t)b * K_ * D_ + i], accum[i]);
    if (tid < K_) atomicAdd(&pops[b * K_ + tid], popsL[tid]);
}

// ---------------- kernel 3: residual + spectral norm + normalize ----------------
// LDS: R @0 (65536), v @65536 (512), w @66048 (512), red @66560 (1024)
__global__ __launch_bounds__(256)
void vlad_finalize(const float* __restrict__ centroids, const float* __restrict__ desc_sums,
                   const float* __restrict__ pops, float* __restrict__ out) {
    extern __shared__ char smem[];
    float* R   = (float*)(smem);
    float* v   = (float*)(smem + 65536);
    float* w   = (float*)(smem + 66048);
    float* red = (float*)(smem + 66560);

    const int b   = blockIdx.x;
    const int tid = threadIdx.x;

    for (int i = tid; i < K_ * D_; i += 256) {
        int k = i >> 7;
        R[i] = centroids[i] * pops[b * K_ + k] - desc_sums[(size_t)b * K_ * D_ + i];
    }
    if (tid < D_) v[tid] = 1.0f + 0.0625f * (float)(tid & 7);
    __syncthreads();

    // Power iteration on R^T R for largest singular value
    for (int it = 0; it < 64; ++it) {
        if (tid < K_) {
            float s = 0.0f;
            for (int d = 0; d < D_; ++d) s += R[tid * D_ + d] * v[d];
            w[tid] = s;
        }
        __syncthreads();
        float u = 0.0f;
        if (tid < D_) {
            for (int k = 0; k < K_; ++k) u += R[k * D_ + tid] * w[k];
        }
        red[tid] = u * u;
        __syncthreads();
        for (int s = 128; s > 0; s >>= 1) {
            if (tid < s) red[tid] += red[tid + s];
            __syncthreads();
        }
        float nrm = sqrtf(red[0]) + 1e-30f;
        if (tid < D_) v[tid] = u / nrm;
        __syncthreads();
    }

    // sigma = ||R v|| with ||v|| = 1
    if (tid < K_) {
        float s = 0.0f;
        for (int d = 0; d < D_; ++d) s += R[tid * D_ + d] * v[d];
        w[tid] = s;
    }
    red[tid] = 0.0f;
    __syncthreads();
    if (tid < K_) red[tid] = w[tid] * w[tid];
    __syncthreads();
    for (int s = 128; s > 0; s >>= 1) {
        if (tid < s) red[tid] += red[tid + s];
        __syncthreads();
    }
    float inv = 1.0f / (sqrtf(red[0]) + 1e-30f);
    for (int i = tid; i < K_ * D_; i += 256)
        out[(size_t)b * K_ * D_ + i] = R[i] * inv;
}

// ---------------- launcher ----------------
extern "C" void kernel_launch(void* const* d_in, const int* in_sizes, int n_in,
                              void* d_out, int out_size, void* d_ws, size_t ws_size,
                              hipStream_t stream) {
    (void)in_sizes; (void)n_in; (void)out_size; (void)ws_size;
    const float* descs         = (const float*)d_in[0];
    const float* centroid_sums = (const float*)d_in[1];
    const float* populations   = (const float*)d_in[2];

    float* ws        = (float*)d_ws;
    float* centroids = ws;                       // 16384
    float* c2        = ws + 16384;               // 128
    float* pops      = ws + 16384 + 128;         // B*K = 2048
    float* desc_sums = pops + B_ * K_;           // B*K*D = 262144
    float* out       = (float*)d_out;

    // pops + desc_sums are contiguous: zero both in one pass
    const int nz = B_ * K_ + B_ * K_ * D_;
    vlad_zero<<<(nz + 255) / 256, 256, 0, stream>>>(pops, nz);

    vlad_prep<<<1, 256, 0, stream>>>(centroid_sums, populations, centroids, c2);

    const size_t smemA = 164864 + 512;           // ~165 KB (fits 320 KB WGP LDS)
    vlad_assign<<<B_ * (N_ / ROWS_PER_BLOCK), 256, smemA, stream>>>(
        descs, centroids, c2, desc_sums, pops);

    const size_t smemF = 66560 + 1024;
    vlad_finalize<<<B_, 256, smemF, stream>>>(centroids, desc_sums, pops, out);
}
